// Attention_18279380811870
// MI455X (gfx1250) — compile-verified, matrix operand order checked by reading, not run on hardware
//
#include <hip/hip_runtime.h>
#include <hip/hip_bf16.h>
#include <math.h>

// Scaled dot-product attention for B=16, L=2048, D=128 (fp32 in/out).
// Outputs: ctx [B,L,D] then probs [B,L,L], concatenated flat in d_out.
//
// MI455X strategy:
//  - f16 WMMA (v_wmma_f32_16x16x32_f16) for S=QK^T and ctx=P*V, f32 accum.
//  - Per-wave 16-query tile; full 16x2048 score strip lives in LDS (f16),
//    exploiting the 320KB WGP LDS so the 256MB probs tensor hits HBM once.
//  - K/V tiles staged with GLOBAL_LOAD_ASYNC_TO_LDS_B128 (ASYNCcnt),
//    double-buffered: issue tile n+1, s_wait_asynccnt ops(n+1), barrier,
//    compute tile n. Fallback: float4 sync copy + global_prefetch.
//  - Row max / row sum via __shfl_xor across the 16 lanes holding one C row.

typedef _Float16 v16h __attribute__((ext_vector_type(16)));
typedef float    v8f  __attribute__((ext_vector_type(8)));
typedef int      v4i  __attribute__((ext_vector_type(4)));

#define B_    16
#define L_    2048
#define D_    128
#define QT    16            // query rows per wave (WMMA M)
#define WAVES 4
#define BLOCK (WAVES * 32)
#define SCALE 0.08838834764831845f   // 1/sqrt(128)

static_assert(BLOCK == 128, "block must be 4 waves of 32");

#if defined(__has_builtin)
# if __has_builtin(__builtin_amdgcn_global_load_async_to_lds_b128) && \
     __has_builtin(__builtin_amdgcn_s_wait_asynccnt)
#  define USE_ASYNC 1
# endif
#endif
#ifndef USE_ASYNC
# define USE_ASYNC 0
#endif

#if USE_ASYNC
typedef __attribute__((address_space(1))) v4i* gbl_v4i_p;   // global src
typedef __attribute__((address_space(3))) v4i* lds_v4i_p;   // LDS dst
#endif

#define STRIP_BYTES ((size_t)WAVES * QT * L_ * 2)   // 256 KB f16 score strips
#define STAGE_FLOATS (32 * D_)                      // one V-sized tile (f32)
#define STAGE_BYTES  ((size_t)2 * STAGE_FLOATS * 4) // double buffered, 32 KB
#define LDS_TOTAL    (STRIP_BYTES + STAGE_BYTES + WAVES * QT * 4)

// K-index striping within a 16-bit A/B fragment chunk (ISA 7.12.2,
// 16-bit A-matrix 16x32): VGPR v<4 -> K = 2v + 8*half + lo,
// VGPR v>=4 -> K = 16 + 2(v-4) + 8*half + lo.
__device__ __forceinline__ int klocal(int e, int hlf) {
    int v = e >> 1, lo = e & 1;
    return (v < 4) ? (2 * v + 8 * hlf + lo) : (16 + 2 * (v - 4) + 8 * hlf + lo);
}

// Stage NFLOATS contiguous f32 from global into LDS, 16B per lane per op.
template <int NFLOATS>
__device__ __forceinline__ void stage_tile(float* lds, const float* gsrc, int tid) {
#pragma unroll
    for (int c = 0; c < NFLOATS / (BLOCK * 4); ++c) {
        int idx = (tid + c * BLOCK) * 4;
#if USE_ASYNC
        __builtin_amdgcn_global_load_async_to_lds_b128(
            (gbl_v4i_p)(gsrc + idx), (lds_v4i_p)(lds + idx), 0, 0);
#else
        *(float4*)(lds + idx) = *(const float4*)(gsrc + idx);
#endif
    }
}

template <int N>
__device__ __forceinline__ void wait_stage() {
#if USE_ASYNC
    __builtin_amdgcn_s_wait_asynccnt(N);
#endif
}

__global__ void attn_fused_kernel(const float* __restrict__ Qg,
                                  const float* __restrict__ Kg,
                                  const float* __restrict__ Vg,
                                  float* __restrict__ ctx,
                                  float* __restrict__ probs) {
    extern __shared__ char smem[];

    const int tid  = threadIdx.x;
    const int lane = tid & 31;
    const int w    = tid >> 5;
    const int hlf  = lane >> 4;     // which 16-lane half
    const int l16  = lane & 15;
    const int b    = blockIdx.y;
    const int q0   = blockIdx.x * (WAVES * QT) + w * QT;

    _Float16* strip = (_Float16*)smem + (size_t)w * QT * L_;
    float*    stg0  = (float*)(smem + STRIP_BYTES);
    float*    stg1  = stg0 + STAGE_FLOATS;
    float*    linv  = (float*)(smem + STRIP_BYTES + STAGE_BYTES) + w * QT;

    // ---- Load Q fragments once (scale folded in): 16x128 as 4x 16x32 f16 ----
    v16h qa[4];
#pragma unroll
    for (int c = 0; c < 4; ++c) {
        const float* qrow = Qg + ((size_t)(b * L_ + q0 + l16)) * D_ + c * 32;
#pragma unroll
        for (int e = 0; e < 16; ++e) {
            qa[c][e] = (_Float16)(qrow[klocal(e, hlf)] * SCALE);
        }
    }

    float m[8], l[8];
#pragma unroll
    for (int r = 0; r < 8; ++r) { m[r] = -1e30f; l[r] = 0.0f; }

    // ---- Sweep 1: S = (Q*scale) K^T -> strip (f16), track row max ----
    {
        const float* Kbase = Kg + (size_t)b * L_ * D_;
        constexpr int KTF = 16 * D_;                 // floats per K tile
        constexpr int KOPS = KTF / (BLOCK * 4);      // async ops/thread/tile
        stage_tile<KTF>(stg0, Kbase, tid);
        for (int kt = 0; kt < L_ / 16; ++kt) {
            float* cur = (kt & 1) ? stg1 : stg0;
            float* nxt = (kt & 1) ? stg0 : stg1;
            if (kt + 1 < L_ / 16) {
#if !USE_ASYNC
                __builtin_prefetch(Kbase + (size_t)(kt + 1) * KTF, 0, 0);
#endif
                stage_tile<KTF>(nxt, Kbase + (size_t)(kt + 1) * KTF, tid);
                wait_stage<KOPS>();                  // tile kt resident
            } else {
                wait_stage<0>();
            }
            __syncthreads();

            v8f cacc = {};
#pragma unroll
            for (int ck = 0; ck < 4; ++ck) {
                v16h bf;                             // B[k][n] = K[key n][dim k]
#pragma unroll
                for (int e = 0; e < 16; ++e) {
                    bf[e] = (_Float16)cur[l16 * D_ + ck * 32 + klocal(e, hlf)];
                }
                cacc = __builtin_amdgcn_wmma_f32_16x16x32_f16(
                    false, qa[ck], false, bf, (short)0, cacc, false, false);
            }
#pragma unroll
            for (int r = 0; r < 8; ++r) {
                m[r] = fmaxf(m[r], cacc[r]);
                strip[(r + 8 * hlf) * L_ + kt * 16 + l16] = (_Float16)cacc[r];
            }
            __syncthreads();                         // all done reading cur
        }
    }

    // Row max across the 16 lanes of each C row (xor stays within each half)
#pragma unroll
    for (int r = 0; r < 8; ++r) {
#pragma unroll
        for (int off = 1; off < 16; off <<= 1) {
            m[r] = fmaxf(m[r], __shfl_xor(m[r], off, 32));
        }
    }

    // ---- Sweep 2: e = exp(s - m) in LDS, accumulate row sums ----
    for (int nt = 0; nt < L_ / 16; ++nt) {
#pragma unroll
        for (int r = 0; r < 8; ++r) {
            int idx = (r + 8 * hlf) * L_ + nt * 16 + l16;
            float e = __expf((float)strip[idx] - m[r]);
            l[r] += e;
            strip[idx] = (_Float16)e;
        }
    }
#pragma unroll
    for (int r = 0; r < 8; ++r) {
#pragma unroll
        for (int off = 1; off < 16; off <<= 1) {
            l[r] += __shfl_xor(l[r], off, 32);
        }
        if (l16 == r) linv[r + 8 * hlf] = 1.0f / l[r];
    }

    // ---- Write normalized probs exactly once (float4, coalesced) ----
    for (int idx = lane * 4; idx < QT * L_; idx += 32 * 4) {
        int row = idx >> 11, col = idx & (L_ - 1);
        float il = linv[row];
        float4 p;
        p.x = (float)strip[idx + 0] * il;
        p.y = (float)strip[idx + 1] * il;
        p.z = (float)strip[idx + 2] * il;
        p.w = (float)strip[idx + 3] * il;
        *(float4*)&probs[((size_t)(b * L_ + q0 + row)) * L_ + col] = p;
    }

    // ---- Sweep 3: ctx = P @ V ----
    v8f acc[8];
#pragma unroll
    for (int t = 0; t < 8; ++t) acc[t] = (v8f){};

    {
        const float* Vbase = Vg + (size_t)b * L_ * D_;
        constexpr int VTF = 32 * D_;                 // floats per V tile
        constexpr int VOPS = VTF / (BLOCK * 4);
        __syncthreads();                             // stage bufs free
        stage_tile<VTF>(stg0, Vbase, tid);
        for (int kt2 = 0; kt2 < L_ / 32; ++kt2) {
            float* cur = (kt2 & 1) ? stg1 : stg0;
            float* nxt = (kt2 & 1) ? stg0 : stg1;
            if (kt2 + 1 < L_ / 32) {
#if !USE_ASYNC
                __builtin_prefetch(Vbase + (size_t)(kt2 + 1) * VTF, 0, 0);
#endif
                stage_tile<VTF>(nxt, Vbase + (size_t)(kt2 + 1) * VTF, tid);
                wait_stage<VOPS>();
            } else {
                wait_stage<0>();
            }
            __syncthreads();

            v16h pa;                                 // A[m][k] = P[q0+m][kt2*32+k]
            float il = linv[l16];
#pragma unroll
            for (int e = 0; e < 16; ++e) {
                pa[e] = (_Float16)((float)strip[l16 * L_ + kt2 * 32 + klocal(e, hlf)] * il);
            }
#pragma unroll
            for (int t = 0; t < 8; ++t) {
                v16h vf;                             // B[k][n] = V[key k][dim t*16+n]
#pragma unroll
                for (int e = 0; e < 16; ++e) {
                    vf[e] = (_Float16)cur[klocal(e, hlf) * D_ + t * 16 + l16];
                }
                acc[t] = __builtin_amdgcn_wmma_f32_16x16x32_f16(
                    false, pa, false, vf, (short)0, acc[t], false, false);
            }
            __syncthreads();
        }
    }

#pragma unroll
    for (int t = 0; t < 8; ++t) {
#pragma unroll
        for (int r = 0; r < 8; ++r) {
            ctx[((size_t)(b * L_ + q0 + r + 8 * hlf)) * D_ + t * 16 + l16] = acc[t][r];
        }
    }
}

extern "C" void kernel_launch(void* const* d_in, const int* in_sizes, int n_in,
                              void* d_out, int out_size, void* d_ws, size_t ws_size,
                              hipStream_t stream) {
    (void)in_sizes; (void)n_in; (void)out_size; (void)d_ws; (void)ws_size;
    const float* q = (const float*)d_in[0];
    const float* k = (const float*)d_in[1];
    const float* v = (const float*)d_in[2];
    float* ctx   = (float*)d_out;
    float* probs = ctx + (size_t)B_ * L_ * D_;

    // ~288.25 KB dynamic LDS (< 320 KB/WGP on CDNA5): raise the cap.
    (void)hipFuncSetAttribute((const void*)attn_fused_kernel,
                              hipFuncAttributeMaxDynamicSharedMemorySize,
                              (int)LDS_TOTAL);

    dim3 grid(L_ / (WAVES * QT), B_);
    attn_fused_kernel<<<grid, dim3(BLOCK), LDS_TOTAL, stream>>>(q, k, v, ctx, probs);
}
